// SetAbstractionLayer_34892314313401
// MI455X (gfx1250) — matrix operand order repaired and checked by reading, not compiled, not run on hardware
//
#include <hip/hip_runtime.h>
#include <math.h>

typedef __attribute__((ext_vector_type(16))) _Float16 v16h;
typedef __attribute__((ext_vector_type(8)))  float    v8f;

// Problem shape (fixed by setup_inputs)
constexpr int kB   = 4;
constexpr int kN   = 8192;
constexpr int kC   = 64;
constexpr int kD   = 128;   // Dout
constexpr int kS   = 2048;  // sample_count
constexpr int kK   = 32;    // neighbors
constexpr int kKin = kC + 3;  // 67
constexpr int kKp  = 96;      // layer-1 K padded to 3 * 32

__device__ __forceinline__ float gelu_exact(float x) {
  return 0.5f * x * (1.0f + erff(x * 0.70710678118654752f));
}

// ---------------------------------------------------------------------------
// Kernel 1: linspace sampling -> sampled coords (also first output section)
// ---------------------------------------------------------------------------
__global__ void sample_kernel(const float* __restrict__ coords,
                              float* __restrict__ out_sampled) {
  int t = blockIdx.x * blockDim.x + threadIdx.x;  // 0 .. B*S-1
  if (t >= kB * kS) return;
  int b = t / kS, s = t % kS;
  const float step = (float)(kN - 1) / (float)(kS - 1);
  int si = (int)rintf((float)s * step);           // round-half-even like jnp.round
  if (si > kN - 1) si = kN - 1;
  const float* src = coords + ((size_t)b * kN + si) * 3;
  float* dst = out_sampled + (size_t)t * 3;
  dst[0] = src[0]; dst[1] = src[1]; dst[2] = src[2];
}

// ---------------------------------------------------------------------------
// Kernel 2: exact kNN (k=32) per sampled point. One 256-thread block / sample.
// Distances live in LDS; 32 rounds of block-argmin with low-index tie-break
// (matches jax.lax.top_k tie behavior). Only the extracted candidate's owner
// rescans its 32 slots per round.
// ---------------------------------------------------------------------------
__global__ void __launch_bounds__(256) knn_kernel(
    const float* __restrict__ coords,
    const float* __restrict__ sampled,
    int* __restrict__ knn) {
  constexpr int PPT = kN / 256;  // points per thread = 32
  __shared__ float dl[kN];       // 32 KB distances
  __shared__ float sv[256];
  __shared__ int   si_[256];

  const int sid = blockIdx.x;    // 0 .. B*S-1
  const int b   = sid / kS;
  const int tid = threadIdx.x;

  const float* cb = coords + (size_t)b * kN * 3;
  const float sx = sampled[sid * 3 + 0];
  const float sy = sampled[sid * 3 + 1];
  const float sz = sampled[sid * 3 + 2];

  // fill distances: thread owns indices c*256 + tid
  #pragma unroll 4
  for (int c = 0; c < PPT; ++c) {
    int i = c * 256 + tid;
    float dx = cb[i * 3 + 0] - sx;
    float dy = cb[i * 3 + 1] - sy;
    float dz = cb[i * 3 + 2] - sz;
    dl[i] = dx * dx + dy * dy + dz * dz;
  }
  __syncthreads();

  // per-thread running min over its candidates (ascending index => stable)
  float mv = dl[tid]; int mi = tid;
  for (int c = 1; c < PPT; ++c) {
    float v = dl[c * 256 + tid];
    if (v < mv) { mv = v; mi = c * 256 + tid; }
  }

  for (int r = 0; r < kK; ++r) {
    sv[tid] = mv; si_[tid] = mi;
    __syncthreads();
    for (int off = 128; off > 0; off >>= 1) {
      if (tid < off) {
        float ov = sv[tid + off]; int oi = si_[tid + off];
        if (ov < sv[tid] || (ov == sv[tid] && oi < si_[tid])) {
          sv[tid] = ov; si_[tid] = oi;
        }
      }
      __syncthreads();
    }
    const int widx = si_[0];
    __syncthreads();
    if (tid == 0) knn[(size_t)sid * kK + r] = widx;
    if ((widx & 255) == tid) {
      dl[widx] = 3.4e38f;
      mv = dl[tid]; mi = tid;
      for (int c = 1; c < PPT; ++c) {
        float v = dl[c * 256 + tid];
        if (v < mv) { mv = v; mi = c * 256 + tid; }
      }
    }
  }
}

// ---------------------------------------------------------------------------
// Kernel 3: per-sample MLP with WMMA f16->f32.
// One wave (32 lanes) per sample; 8 waves / 256-thread block.
// Weights staged into LDS transposed (column-major) so each B fragment is a
// contiguous 32B-aligned v16h LDS load. h (32x128) bridges the C/D->A lane
// transpose through a per-wave LDS region.
// ---------------------------------------------------------------------------
__global__ void __launch_bounds__(256) mlp_kernel(
    const float* __restrict__ coords,
    const float* __restrict__ features,
    const float* __restrict__ W1,
    const float* __restrict__ b1,
    const float* __restrict__ W2,
    const float* __restrict__ b2,
    const float* __restrict__ sampled,
    const int* __restrict__ knn,
    float* __restrict__ pooled) {
  __shared__ alignas(32) _Float16 W1sT[kD * kKp];    // [n][k], 24 KB
  __shared__ alignas(32) _Float16 W2sT[kD * kD];     // [n][k], 32 KB
  __shared__ alignas(32) _Float16 Hs[8][kK * kD];    // per-wave h, 64 KB

  const int tid = threadIdx.x;

  // cooperative transposed weight staging (f32 -> f16), zero-pad K 67->96
  for (int i = tid; i < kD * kKp; i += 256) {
    int nc = i / kKp, kr = i % kKp;
    W1sT[i] = (kr < kKin) ? (_Float16)W1[kr * kD + nc] : (_Float16)0.f;
  }
  for (int i = tid; i < kD * kD; i += 256) {
    int nc = i / kD, kr = i % kD;
    W2sT[i] = (_Float16)W2[kr * kD + nc];
  }
  __syncthreads();

  const int wv   = tid >> 5;
  const int lane = tid & 31;
  const int half = lane >> 4;      // 0: lanes 0-15, 1: lanes 16-31
  const int col  = lane & 15;
  const int sid  = blockIdx.x * 8 + wv;   // 0 .. B*S-1
  const int b    = sid / kS;

  // A-matrix rows this lane supplies: m0 (tile 0) and m1 (tile 1)
  const int m0 = col, m1 = 16 + col;
  const int i0 = knn[(size_t)sid * kK + m0];
  const int i1 = knn[(size_t)sid * kK + m1];
  const float sx = sampled[sid * 3 + 0];
  const float sy = sampled[sid * 3 + 1];
  const float sz = sampled[sid * 3 + 2];
  const float* cb = coords   + (size_t)b * kN * 3;
  const float* fb = features + (size_t)b * kN * kC;
  const float rel0[3] = { cb[i0 * 3 + 0] - sx, cb[i0 * 3 + 1] - sy, cb[i0 * 3 + 2] - sz };
  const float rel1[3] = { cb[i1 * 3 + 0] - sx, cb[i1 * 3 + 1] - sy, cb[i1 * 3 + 2] - sz };
  const float* f0 = fb + (size_t)i0 * kC;
  const float* f1 = fb + (size_t)i1 * kC;

  v8f acc[2][8];
  #pragma unroll
  for (int mt = 0; mt < 2; ++mt)
    #pragma unroll
    for (int nt = 0; nt < 8; ++nt)
      #pragma unroll
      for (int r = 0; r < 8; ++r) acc[mt][nt][r] = 0.f;

  // ---------------- layer 1: G(32x67,pad96) @ W1 -> 32x128 ----------------
  #pragma unroll
  for (int kt = 0; kt < 3; ++kt) {
    const int kbase = kt * 32;
    v16h a0, a1;
    #pragma unroll
    for (int e = 0; e < 16; ++e) {
      // A k order chosen here (contiguous, same as B): koff = e + 16*half
      const int kk = kbase + e + 16 * half;
      float g0, g1;
      if (kk < 3)         { g0 = rel0[kk];     g1 = rel1[kk]; }
      else if (kk < kKin) { g0 = f0[kk - 3];   g1 = f1[kk - 3]; }
      else                { g0 = 0.f;          g1 = 0.f; }
      a0[e] = (_Float16)g0;
      a1[e] = (_Float16)g1;
    }
    #pragma unroll
    for (int nt = 0; nt < 8; ++nt) {
      // B k order chosen here (contiguous, same as A): element e -> K = kbase + e + 16*half, N = col
      const v16h bv = *(const v16h*)(W1sT + (size_t)(nt * 16 + col) * kKp + kbase + 16 * half);
      acc[0][nt] = __builtin_amdgcn_wmma_f32_16x16x32_f16(
          false, a0, false, bv, (short)0, acc[0][nt], false, false);
      acc[1][nt] = __builtin_amdgcn_wmma_f32_16x16x32_f16(
          false, a1, false, bv, (short)0, acc[1][nt], false, false);
    }
  }

  // bias + gelu, spill h (f16) into per-wave LDS region
  _Float16* hrow = Hs[wv];
  #pragma unroll
  for (int nt = 0; nt < 8; ++nt) {
    const float bias = b1[nt * 16 + col];
    #pragma unroll
    for (int mt = 0; mt < 2; ++mt) {
      #pragma unroll
      for (int r = 0; r < 8; ++r) {
        // C/D layout: VGPR r of tile mt, this lane = row mt*16 + half*8 + r
        const int M = mt * 16 + half * 8 + r;
        hrow[M * kD + nt * 16 + col] = (_Float16)gelu_exact(acc[mt][nt][r] + bias);
      }
    }
  }
  __syncthreads();  // uniform across block; orders per-wave LDS RAW too

  // ---------------- layer 2: h(32x128) @ W2 -> 32x128 ----------------
  #pragma unroll
  for (int mt = 0; mt < 2; ++mt)
    #pragma unroll
    for (int nt = 0; nt < 8; ++nt)
      #pragma unroll
      for (int r = 0; r < 8; ++r) acc[mt][nt][r] = 0.f;

  #pragma unroll
  for (int kt = 0; kt < 4; ++kt) {
    const int kbase = kt * 32;
    v16h a0, a1;
    #pragma unroll
    for (int e = 0; e < 16; ++e) {
      const int kk = kbase + e + 16 * half;
      a0[e] = hrow[m0 * kD + kk];
      a1[e] = hrow[m1 * kD + kk];
    }
    #pragma unroll
    for (int nt = 0; nt < 8; ++nt) {
      const v16h bv = *(const v16h*)(W2sT + (size_t)(nt * 16 + col) * kD + kbase + 16 * half);
      acc[0][nt] = __builtin_amdgcn_wmma_f32_16x16x32_f16(
          false, a0, false, bv, (short)0, acc[0][nt], false, false);
      acc[1][nt] = __builtin_amdgcn_wmma_f32_16x16x32_f16(
          false, a1, false, bv, (short)0, acc[1][nt], false, false);
    }
  }

  // bias + gelu + max over the 32 neighbors (= M dimension), then store
  #pragma unroll
  for (int nt = 0; nt < 8; ++nt) {
    const float bias = b2[nt * 16 + col];
    float m = -3.4e38f;
    #pragma unroll
    for (int mt = 0; mt < 2; ++mt)
      #pragma unroll
      for (int r = 0; r < 8; ++r)
        m = fmaxf(m, gelu_exact(acc[mt][nt][r] + bias));
    // lane L holds rows {..} for column nt*16+col; merge the two lane halves
    m = fmaxf(m, __shfl_xor(m, 16, 32));
    if (half == 0)
      pooled[(size_t)sid * kD + nt * 16 + col] = m;
  }
}

// ---------------------------------------------------------------------------
extern "C" void kernel_launch(void* const* d_in, const int* in_sizes, int n_in,
                              void* d_out, int out_size, void* d_ws, size_t ws_size,
                              hipStream_t stream) {
  const float* coords   = (const float*)d_in[0];
  const float* features = (const float*)d_in[1];
  const float* W1 = (const float*)d_in[2];
  const float* b1 = (const float*)d_in[3];
  const float* W2 = (const float*)d_in[4];
  const float* b2 = (const float*)d_in[5];
  // d_in[6] = sample_count (2048), d_in[7] = k (32): fixed by setup, hardcoded.

  float* out         = (float*)d_out;
  float* out_sampled = out;                          // B*S*3 floats
  float* out_pooled  = out + (size_t)kB * kS * 3;    // B*S*D floats
  int*   knn         = (int*)d_ws;                   // B*S*K ints (1 MB)

  sample_kernel<<<(kB * kS + 255) / 256, 256, 0, stream>>>(coords, out_sampled);
  knn_kernel<<<kB * kS, 256, 0, stream>>>(coords, out_sampled, knn);
  mlp_kernel<<<(kB * kS) / 8, 256, 0, stream>>>(coords, features, W1, b1, W2, b2,
                                                out_sampled, knn, out_pooled);
}
